// VectorQuantizer_41583873360723
// MI455X (gfx1250) — matrix operand order repaired and checked by reading, not compiled, not run on hardware
//
#include <hip/hip_runtime.h>
#include <hip/hip_bf16.h>

typedef __attribute__((ext_vector_type(16))) __bf16 v16bf;
typedef __attribute__((ext_vector_type(8)))  float  v8f;

#define C_DIM   128
#define HW      1024            // 32*32
#define K_CODES 256
#define N_ROWS  131072          // 128 * 1024
#define Q_ELEMS 16777216        // 128*128*32*32
#define IDX_OFF Q_ELEMS
#define LOSS_OFF (Q_ELEMS + N_ROWS)

#define EB_STRIDE 136           // bf16 elements per emb row in LDS (pad 8)
#define ZS_STRIDE 132           // f32 elements per z row in LDS (pad 4)

__global__ __launch_bounds__(256) void vq_main(const float* __restrict__ z,
                                               const float* __restrict__ emb,
                                               float* __restrict__ out,
                                               float* __restrict__ partial)
{
    __shared__ __align__(16) __bf16 eb[K_CODES * EB_STRIDE];
    __shared__ float e2[K_CODES];
    __shared__ __align__(16) float zs[8 * 16 * ZS_STRIDE];
    __shared__ float wsum[8];

    const int tid = threadIdx.x;
    const int wg  = blockIdx.x;

    // ---- Phase 0: stage embedding as bf16 + squared norms into LDS ----
    {
        const int code = tid;                       // 256 threads <-> 256 codes
        const float4* er = (const float4*)(emb + code * C_DIM);
        float s = 0.f;
        #pragma unroll 4
        for (int i = 0; i < C_DIM / 4; ++i) {
            float4 v = er[i];
            s += v.x * v.x + v.y * v.y + v.z * v.z + v.w * v.w;
            __bf16* dst = &eb[code * EB_STRIDE + i * 4];
            dst[0] = (__bf16)v.x; dst[1] = (__bf16)v.y;
            dst[2] = (__bf16)v.z; dst[3] = (__bf16)v.w;
        }
        e2[code] = s;
    }
    __syncthreads();

    const int w    = tid >> 5;          // wave 0..7
    const int lane = tid & 31;
    const int l16  = lane & 15;
    const int g    = lane >> 4;

    const int n0   = wg * 128 + w * 16;         // first row of this wave's tile
    const int bi   = n0 >> 10;                  // batch index (HW = 1024)
    const int hw0  = n0 & 1023;                 // tiles never straddle a batch
    const size_t zbase = (size_t)bi * (C_DIM * HW) + hw0;

    float* zw = &zs[w * 16 * ZS_STRIDE];

    // ---- Phase 1: stage this wave's 16 z rows (f32) into LDS ----
    // lanes 0-15 handle even c, lanes 16-31 odd c; each half-wave reads a
    // 64B contiguous segment (16 consecutive hw).
    for (int it = 0; it < 64; ++it) {
        int c = it * 2 + g;
        float v = z[zbase + (size_t)c * HW + l16];
        zw[l16 * ZS_STRIDE + c] = v;
    }
    __syncthreads();

    // ---- Phase 2: build the 4 B fragments (z tile, K-steps of 32) ----
    v16bf bfrag[4];
    #pragma unroll
    for (int s = 0; s < 4; ++s) {
        const float4* src = (const float4*)&zw[l16 * ZS_STRIDE + s * 32 + g * 8];
        float tmp[16];
        *(float4*)&tmp[0]  = src[0];
        *(float4*)&tmp[4]  = src[1];
        *(float4*)&tmp[8]  = src[4];
        *(float4*)&tmp[12] = src[5];
        v16bf bv;
        #pragma unroll
        for (int e = 0; e < 16; ++e) bv[e] = (__bf16)tmp[e];
        bfrag[s] = bv;
    }

    // ---- Phase 3: 16 code tiles x 4 K-steps of WMMA, running argmin ----
    float best  = 3.4e38f;
    int   bcode = 0;
    for (int t = 0; t < 16; ++t) {
        v8f acc = {};
        const int code = t * 16 + l16;          // A-matrix row M = lane%16
        #pragma unroll
        for (int s = 0; s < 4; ++s) {
            union { v16bf v; uint4 q[2]; } a;
            // elements 0..7 : c = s*32 + g*8 + {0..7}
            a.q[0] = *(const uint4*)&eb[code * EB_STRIDE + s * 32 + g * 8];
            // elements 8..15: c = s*32 + 16 + g*8 + {0..7}
            a.q[1] = *(const uint4*)&eb[code * EB_STRIDE + s * 32 + 16 + g * 8];
            acc = __builtin_amdgcn_wmma_f32_16x16x32_bf16(
                false, a.v, false, bfrag[s], (short)0, acc, false, false);
        }
        // D: VGPR r holds code m = t*16 + 8*g + r for column n = lane%16
        #pragma unroll
        for (int r = 0; r < 8; ++r) {
            int   m     = t * 16 + g * 8 + r;
            float score = e2[m] - 2.0f * acc[r];   // ||e||^2 - 2 z.e (z^2 const)
            if (score < best) { best = score; bcode = m; }   // ascending m => lowest-index tie-break
        }
    }
    // merge the two lane halves (each covers half the codes for column n)
    {
        float ob = __shfl_xor(best, 16, 32);
        int   oi = __shfl_xor(bcode, 16, 32);
        if (ob < best || (ob == best && oi < bcode)) { best = ob; bcode = oi; }
    }

    // ---- Phase 4: outputs ----
    if (g == 0) out[IDX_OFF + n0 + l16] = (float)bcode;   // indices as float

    const float* qrow = emb + bcode * C_DIM;              // exact f32 codebook row (L2-hot)
    float lsum = 0.f;
    for (int it = 0; it < 64; ++it) {
        int   c  = it * 2 + g;
        float q  = qrow[c];
        float zv = zw[l16 * ZS_STRIDE + c];
        float d  = q - zv;
        lsum += d * d;
        out[zbase + (size_t)c * HW + l16] = q;            // quantized_st == quantized
    }

    // deterministic wave -> workgroup reduction of the loss partial
    #pragma unroll
    for (int off = 16; off > 0; off >>= 1) lsum += __shfl_xor(lsum, off, 32);
    if (lane == 0) wsum[w] = lsum;
    __syncthreads();
    if (tid == 0) {
        float s = 0.f;
        #pragma unroll
        for (int i = 0; i < 8; ++i) s += wsum[i];
        partial[wg] = s;
    }
}

__global__ __launch_bounds__(256) void vq_reduce(const float* __restrict__ partial,
                                                 float* __restrict__ out)
{
    __shared__ float sm[256];
    const int t = threadIdx.x;
    float s = partial[t] + partial[t + 256] + partial[t + 512] + partial[t + 768];
    sm[t] = s;
    __syncthreads();
    for (int off = 128; off > 0; off >>= 1) {
        if (t < off) sm[t] += sm[t + off];
        __syncthreads();
    }
    if (t == 0) {
        float mean = sm[0] / (float)Q_ELEMS;
        out[LOSS_OFF]     = 0.25f * mean;   // commitment loss
        out[LOSS_OFF + 1] = mean;           // codebook loss
    }
}

extern "C" void kernel_launch(void* const* d_in, const int* in_sizes, int n_in,
                              void* d_out, int out_size, void* d_ws, size_t ws_size,
                              hipStream_t stream)
{
    const float* z   = (const float*)d_in[0];   // [128,128,32,32] f32
    const float* emb = (const float*)d_in[1];   // [256,128] f32
    float* out       = (float*)d_out;           // quantized | indices | 2 losses
    float* partial   = (float*)d_ws;            // 1024 f32 partial sums

    vq_main<<<dim3(N_ROWS / 128), dim3(256), 0, stream>>>(z, emb, out, partial);
    vq_reduce<<<dim3(1), dim3(256), 0, stream>>>(partial, out);
}
